// Decodeur_67147518705910
// MI455X (gfx1250) — matrix operand order repaired and checked by reading, not compiled, not run on hardware
//
#include <hip/hip_runtime.h>
#include <hip/hip_bf16.h>

typedef __attribute__((ext_vector_type(16))) _Float16 v16h;
typedef __attribute__((ext_vector_type(8)))  float    v8f;
typedef __attribute__((ext_vector_type(4)))  float    v4f;

#define UNITS   80
#define LATENT  128
#define IN_DIM  256
#define BATCH   8192
#define TSTEPS  256
#define ROWSTRIDE (TSTEPS * UNITS)   // 20480 floats per output batch row

__device__ __forceinline__ float sigf(float x) { return 1.0f / (1.0f + __expf(-x)); }

// ---- WMMA A fragment loader (wave32; contiguous k order, same as packed B) ---------------------
// A 16x32 f16 (MxK), row-major source, ldA in elements.
// lane 0-15: M=lane, VGPR0..7 = K 0..15
// lane16-31: M=lane-16, VGPR0..7 = K 16..31
__device__ __forceinline__ v16h load_a16(const _Float16* __restrict__ A, int ldA,
                                         int m0, int k0, int lane) {
  const int m    = m0 + (lane & 15);
  const int half = lane >> 4;
  const _Float16* p = A + (size_t)m * ldA + k0 + half * 16;
  v16h a;
#pragma unroll
  for (int e = 0; e < 8; ++e) a[e] = p[e];         // 16B contiguous -> b128
#pragma unroll
  for (int e = 0; e < 8; ++e) a[8 + e] = p[8 + e]; // 16B contiguous -> b128
  return a;
}

// ---- Packed-B fragment loader: one contiguous 32B load per lane ------------
// Packed layout: block (kt*ntiles + nt) holds 512 f16 = [lane 0..31][e 0..15],
// where lane = half*16 + (n&15), e = k&15, half = (k&31)>>4.
__device__ __forceinline__ v16h load_b_packed(const _Float16* __restrict__ P,
                                              int ntiles, int kt, int nt, int lane) {
  const _Float16* p = P + ((size_t)(kt * ntiles + nt) * 512) + lane * 16;
  return *(const v16h*)p;
}

// ---- fp32 -> fp16 convert (for x) -----------------------------------------
__global__ void k_f32_to_f16(const float* __restrict__ s, _Float16* __restrict__ d, int n) {
  int i = blockIdx.x * blockDim.x + threadIdx.x;
  if (i < n) d[i] = (_Float16)s[i];
}

// ---- convert + swizzle a row-major (K x N) f32 weight into B-fragment blocks
__global__ void k_packB(const float* __restrict__ W, _Float16* __restrict__ P,
                        int K, int N) {
  int idx = blockIdx.x * blockDim.x + threadIdx.x;
  if (idx >= K * N) return;
  const int k = idx / N, n = idx - (idx / N) * N;
  const int kt = k >> 5, rem = k & 31, half = rem >> 4, e = rem & 15;
  const int nt = n >> 4, nn = n & 15;
  const int ntiles = N >> 4;
  P[((size_t)(kt * ntiles + nt) * 512) + (half * 16 + nn) * 16 + e] = (_Float16)W[idx];
}

// ---- GEMM1: lat = x(8192x256) @ W(256x128) + b, store f16 ------------------
__global__ void k_gemm_lat(const _Float16* __restrict__ xh, const _Float16* __restrict__ Wp,
                           const float* __restrict__ bias, _Float16* __restrict__ latH) {
  const int lane = threadIdx.x & 31;
  const int wave = threadIdx.x >> 5;
  const int m0 = (blockIdx.x * 4 + wave) * 16;
  const int nt = blockIdx.y;                       // 8 n-tiles
  v8f acc = {};
#pragma unroll
  for (int kt = 0; kt < IN_DIM / 32; ++kt) {
    v16h a = load_a16(xh, IN_DIM, m0, kt * 32, lane);
    v16h b = load_b_packed(Wp, LATENT / 16, kt, nt, lane);
    acc = __builtin_amdgcn_wmma_f32_16x16x32_f16(false, a, false, b, (short)0, acc, false, false);
  }
  const int half = lane >> 4;
  const int n = nt * 16 + (lane & 15);
  const float bn = bias[n];
#pragma unroll
  for (int r = 0; r < 8; ++r) {
    const int m = m0 + half * 8 + r;
    latH[(size_t)m * LATENT + n] = (_Float16)(acc[r] + bn);
  }
}

// ---- GEMM2 (fused gates): h = [gates(lat@Kfw) | gates(lat@Kbw)], store f16 -
// One wave: 16 rows x 16 h-cols; 3 accumulators (i,c,o z-tiles), shared A frag.
__global__ void k_gemm_h(const _Float16* __restrict__ latH,
                         const _Float16* __restrict__ KfwP, const _Float16* __restrict__ KbwP,
                         const float* __restrict__ bfw, const float* __restrict__ bbw,
                         _Float16* __restrict__ hH) {
  const int lane = threadIdx.x & 31;
  const int wave = threadIdx.x >> 5;
  const int m0  = (blockIdx.x * 4 + wave) * 16;
  const int ut  = blockIdx.y;                 // 5 column tiles within [0,80)
  const int dir = blockIdx.z;                 // 0 = forward, 1 = backward
  const _Float16* Kp = dir ? KbwP : KfwP;
  const float* bias  = dir ? bbw : bfw;
  const int NT = (4 * UNITS) / 16;            // 20 n-tiles in z
  v8f ai = {}, ac = {}, ao = {};
#pragma unroll
  for (int kt = 0; kt < LATENT / 32; ++kt) {
    v16h a  = load_a16(latH, LATENT, m0, kt * 32, lane);
    v16h bi = load_b_packed(Kp, NT, kt, ut, lane);       // i gate tile
    v16h bc = load_b_packed(Kp, NT, kt, 10 + ut, lane);  // c~ tile (cols 160+)
    v16h bo = load_b_packed(Kp, NT, kt, 15 + ut, lane);  // o gate tile (cols 240+)
    ai = __builtin_amdgcn_wmma_f32_16x16x32_f16(false, a, false, bi, (short)0, ai, false, false);
    ac = __builtin_amdgcn_wmma_f32_16x16x32_f16(false, a, false, bc, (short)0, ac, false, false);
    ao = __builtin_amdgcn_wmma_f32_16x16x32_f16(false, a, false, bo, (short)0, ao, false, false);
  }
  const int half = lane >> 4;
  const int col = ut * 16 + (lane & 15);
  const float bi_ = bias[col], bc_ = bias[2 * UNITS + col], bo_ = bias[3 * UNITS + col];
#pragma unroll
  for (int r = 0; r < 8; ++r) {
    const int m = m0 + half * 8 + r;
    const float zi = ai[r] + bi_, zc = ac[r] + bc_, zo = ao[r] + bo_;
    // c = sig(zi)*relu(zc) >= 0  =>  h = sig(zo)*sig(zi)*relu(zc)
    const float h = sigf(zo) * sigf(zi) * fmaxf(zc, 0.0f);
    hH[(size_t)m * (2 * UNITS) + dir * UNITS + col] = (_Float16)h;
  }
}

// ---- GEMM3 (fused gates): y0 = gates(h@K1), write into t=0 slots of out ----
__global__ void k_gemm_y0(const _Float16* __restrict__ hH, const _Float16* __restrict__ K1P,
                          const float* __restrict__ b1, float* __restrict__ out) {
  const int lane = threadIdx.x & 31;
  const int wave = threadIdx.x >> 5;
  const int m0 = (blockIdx.x * 4 + wave) * 16;
  const int ut = blockIdx.y;                  // 5 tiles
  const int NT = (4 * UNITS) / 16;            // 20
  v8f ai = {}, ac = {}, ao = {};
#pragma unroll
  for (int kt = 0; kt < (2 * UNITS) / 32; ++kt) {   // K = 160, 5 steps
    v16h a  = load_a16(hH, 2 * UNITS, m0, kt * 32, lane);
    v16h bi = load_b_packed(K1P, NT, kt, ut, lane);
    v16h bc = load_b_packed(K1P, NT, kt, 10 + ut, lane);
    v16h bo = load_b_packed(K1P, NT, kt, 15 + ut, lane);
    ai = __builtin_amdgcn_wmma_f32_16x16x32_f16(false, a, false, bi, (short)0, ai, false, false);
    ac = __builtin_amdgcn_wmma_f32_16x16x32_f16(false, a, false, bc, (short)0, ac, false, false);
    ao = __builtin_amdgcn_wmma_f32_16x16x32_f16(false, a, false, bo, (short)0, ao, false, false);
  }
  const int half = lane >> 4;
  const int col = ut * 16 + (lane & 15);
  const float bi_ = b1[col], bc_ = b1[2 * UNITS + col], bo_ = b1[3 * UNITS + col];
#pragma unroll
  for (int r = 0; r < 8; ++r) {
    const int m = m0 + half * 8 + r;
    const float zi = ai[r] + bi_, zc = ac[r] + bc_, zo = ao[r] + bo_;
    out[(size_t)m * ROWSTRIDE + col] = sigf(zo) * sigf(zi) * fmaxf(zc, 0.0f);
  }
}

// ---- yc: decode_step(ones) — identical for every batch row; exact fp32 -----
__global__ void k_yc(const float* __restrict__ Kfw, const float* __restrict__ bfw,
                     const float* __restrict__ Kbw, const float* __restrict__ bbw,
                     const float* __restrict__ K1,  const float* __restrict__ b1,
                     float* __restrict__ yc) {
  __shared__ float zf[4 * UNITS], zb[4 * UNITS], h1[2 * UNITS], z1[4 * UNITS];
  const int j = threadIdx.x;  // 0..319
  {
    float sf = bfw[j], sb = bbw[j];
    for (int k = 0; k < LATENT; ++k) {
      sf += Kfw[(size_t)k * 4 * UNITS + j];
      sb += Kbw[(size_t)k * 4 * UNITS + j];
    }
    zf[j] = sf; zb[j] = sb;
  }
  __syncthreads();
  if (j < 2 * UNITS) {
    const float* z = (j < UNITS) ? zf : zb;
    const int u = (j < UNITS) ? j : j - UNITS;
    h1[j] = sigf(z[3 * UNITS + u]) * sigf(z[u]) * fmaxf(z[2 * UNITS + u], 0.0f);
  }
  __syncthreads();
  {
    float s = b1[j];
    for (int k = 0; k < 2 * UNITS; ++k) s += h1[k] * K1[(size_t)k * 4 * UNITS + j];
    z1[j] = s;
  }
  __syncthreads();
  if (j < UNITS)
    yc[j] = sigf(z1[3 * UNITS + j]) * sigf(z1[j]) * fmaxf(z1[2 * UNITS + j], 0.0f);
}

// ---- broadcast fill: out[b, 1..255, :] = yc --------------------------------
// 320 threads: 320 % 20 == 0, so each thread's float4 slot within the 80-col
// row is loop-invariant -> value held in registers, pure NT b128 store stream.
__global__ void k_fill(const float* __restrict__ yc, float* __restrict__ out) {
  const int b   = blockIdx.x;
  const int tid = threadIdx.x;                 // 0..319
  const v4f* ycv = (const v4f*)yc;             // 20 x float4
  const v4f val = ycv[tid % 20];
  v4f* o = (v4f*)(out + (size_t)b * ROWSTRIDE);  // 5120 x float4 per batch row
  for (int j = 20 + tid; j < TSTEPS * (UNITS / 4); j += 320)
    __builtin_nontemporal_store(val, &o[j]);
}

extern "C" void kernel_launch(void* const* d_in, const int* in_sizes, int n_in,
                              void* d_out, int out_size, void* d_ws, size_t ws_size,
                              hipStream_t stream) {
  const float* x    = (const float*)d_in[0];
  // d_in[1] = size (always 256 per reference constants)
  const float* Wlat = (const float*)d_in[2];
  const float* blat = (const float*)d_in[3];
  const float* Kfw  = (const float*)d_in[4];
  const float* bfw  = (const float*)d_in[5];
  const float* Kbw  = (const float*)d_in[6];
  const float* bbw  = (const float*)d_in[7];
  const float* K1   = (const float*)d_in[8];
  const float* b1   = (const float*)d_in[9];
  float* out = (float*)d_out;

  char* ws = (char*)d_ws;                       // all offsets 256B aligned
  _Float16* xh    = (_Float16*)(ws + 0);        // 8192*256*2  = 4,194,304
  _Float16* WlatP = (_Float16*)(ws + 4194304);  // 256*128*2   =    65,536
  _Float16* KfwP  = (_Float16*)(ws + 4259840);  // 128*320*2   =    81,920
  _Float16* KbwP  = (_Float16*)(ws + 4341760);  // 128*320*2   =    81,920
  _Float16* K1P   = (_Float16*)(ws + 4423680);  // 160*320*2   =   102,400
  _Float16* latH  = (_Float16*)(ws + 4526080);  // 8192*128*2  = 2,097,152
  _Float16* hH    = (_Float16*)(ws + 6623232);  // 8192*160*2  = 2,621,440
  float*    yc    = (float*)   (ws + 9244672);  // 80*4        =       320

  k_f32_to_f16<<<(BATCH * IN_DIM + 255) / 256, 256, 0, stream>>>(x, xh, BATCH * IN_DIM);
  k_packB<<<(IN_DIM * LATENT + 255) / 256, 256, 0, stream>>>(Wlat, WlatP, IN_DIM, LATENT);
  k_packB<<<(LATENT * 4 * UNITS + 255) / 256, 256, 0, stream>>>(Kfw, KfwP, LATENT, 4 * UNITS);
  k_packB<<<(LATENT * 4 * UNITS + 255) / 256, 256, 0, stream>>>(Kbw, KbwP, LATENT, 4 * UNITS);
  k_packB<<<(2 * UNITS * 4 * UNITS + 255) / 256, 256, 0, stream>>>(K1, K1P, 2 * UNITS, 4 * UNITS);

  k_gemm_lat<<<dim3(BATCH / 64, LATENT / 16), 128, 0, stream>>>(xh, WlatP, blat, latH);
  k_gemm_h  <<<dim3(BATCH / 64, UNITS / 16, 2), 128, 0, stream>>>(latH, KfwP, KbwP, bfw, bbw, hH);
  k_gemm_y0 <<<dim3(BATCH / 64, UNITS / 16), 128, 0, stream>>>(hH, K1P, b1, out);
  k_yc      <<<1, 4 * UNITS, 0, stream>>>(Kfw, bfw, Kbw, bbw, K1, b1, yc);
  k_fill    <<<BATCH, 320, 0, stream>>>(yc, out);
}
